// AttentionHead_26018911879697
// MI455X (gfx1250) — matrix operand order repaired and checked by reading, not compiled, not run on hardware
//
#include <hip/hip_runtime.h>
#include <math.h>

typedef __bf16 bf16_t;
typedef __attribute__((ext_vector_type(16))) __bf16 v16bf;
typedef __attribute__((ext_vector_type(8)))  __bf16 v8bf;
typedef __attribute__((ext_vector_type(8)))  float  v8f;

typedef unsigned int       u32;
typedef unsigned long long u64;
typedef __attribute__((ext_vector_type(4))) u32 u32x4;
typedef __attribute__((ext_vector_type(4))) int i32x4;
typedef __attribute__((ext_vector_type(8))) int i32x8;

#define B_ 4
#define S_ 2048
#define E_ 1024
#define D_ 128
#define M_ (B_ * S_)   // 8192 total rows

// ---------------------------------------------------------------------------
// Tensor Data Mover: 2-D tile load, global -> LDS (ISA 08_async_tensor §8).
// D# group0: count=1 | lds_addr | global_addr[56:0] | type=2
// D# group1: data_size=2B | tensor_dim0/1 | tile_dim0/1 | tensor_dim0_stride
// groups 2/3 zero (tile_dim2=0 -> unused, 2-D tensor).
// ---------------------------------------------------------------------------
__device__ __forceinline__ void tdm_load_2d(u64 gaddr, u32 lds_addr,
                                            u32 dim0, u32 dim1, u64 stride0,
                                            u32 tile0, u32 tile1) {
  u32x4 g0;
  g0[0] = 1u;                                        // count=1, user descriptor
  g0[1] = lds_addr;                                  // LDS byte address
  g0[2] = (u32)(gaddr & 0xFFFFFFFFu);                // global_addr[31:0]
  g0[3] = (u32)((gaddr >> 32) & 0x01FFFFFFu) | (2u << 30);  // [56:32] | type=2
  i32x8 g1;
  g1[0] = (int)(1u << 16);                           // data_size code 1 = 2 bytes
  g1[1] = (int)((dim0 & 0xFFFFu) << 16);             // tensor_dim0[15:0] @ bit48
  g1[2] = (int)((dim0 >> 16) | ((dim1 & 0xFFFFu) << 16));
  g1[3] = (int)((dim1 >> 16) | (tile0 << 16));       // tile_dim0 @ bit112
  g1[4] = (int)(tile1 & 0xFFFFu);                    // tile_dim1 (tile_dim2=0)
  g1[5] = (int)(u32)(stride0 & 0xFFFFFFFFu);         // tensor_dim0_stride[31:0]
  g1[6] = (int)(u32)((stride0 >> 32) & 0xFFFFu);     // stride[47:32], dim1_stride=0
  g1[7] = 0;
  i32x4 gz = {0, 0, 0, 0};
#if defined(__clang_major__) && (__clang_major__ >= 23)
  i32x8 gz8 = {0, 0, 0, 0, 0, 0, 0, 0};
  __builtin_amdgcn_tensor_load_to_lds(g0, g1, gz, gz, gz8, 0);
#else
  __builtin_amdgcn_tensor_load_to_lds(g0, g1, gz, gz, 0);
#endif
}

// ---------------------------------------------------------------------------
// Kernel 1: fused QKV projection  (X[M,E] f32  x  W[E,D] f32) -> bf16
//   z==0 -> Q [M,D] row-major      (A-operand friendly for QK^T)
//   z==1 -> K [M,D] row-major      (B-operand friendly for QK^T)
//   z==2 -> Vt [B][D][S]           (B-operand friendly for P*V)
// ---------------------------------------------------------------------------
__launch_bounds__(256)
__global__ void qkv_proj(const float* __restrict__ X,
                         const float* __restrict__ Wq,
                         const float* __restrict__ Wk,
                         const float* __restrict__ Wv,
                         bf16_t* __restrict__ Q,
                         bf16_t* __restrict__ Kb,
                         bf16_t* __restrict__ Vt) {
  constexpr int XS = 40;                       // padded LDS row stride (bf16 elems)
  __shared__ __align__(32) bf16_t Xs[128 * XS];   // X tile, bf16 [mrow][e]
  __shared__ __align__(32) bf16_t Ws[128 * XS];   // W tile transposed [n][e]

  const int tid  = threadIdx.x;
  const int wave = tid >> 5;
  const int lane = tid & 31;
  const int lrow = lane & 15;
  const int g    = lane >> 4;
  const int m0   = blockIdx.x * 128;
  const int z    = blockIdx.z;
  const float* W = (z == 0) ? Wq : (z == 1) ? Wk : Wv;

  v8f acc[8] = {};

  for (int e0 = 0; e0 < E_; e0 += 32) {
    // Cooperative stage: X[m0..m0+127][e0..e0+31] -> Xs (f32 -> bf16)
    for (int i = 0; i < 4; ++i) {
      int f4  = tid + 256 * i;                 // float4 index, 0..1023
      int row = f4 >> 3;
      int col = (f4 & 7) << 2;
      const float4 xv = *(const float4*)&X[(size_t)(m0 + row) * E_ + e0 + col];
      bf16_t* p = &Xs[row * XS + col];
      p[0] = (bf16_t)xv.x; p[1] = (bf16_t)xv.y;
      p[2] = (bf16_t)xv.z; p[3] = (bf16_t)xv.w;
    }
    // Cooperative stage: W[e0..e0+31][0..127] -> Ws transposed [n][e]
    for (int i = 0; i < 4; ++i) {
      int f4 = tid + 256 * i;
      int er = f4 >> 5;                        // 0..31
      int c  = (f4 & 31) << 2;                 // 0..124
      const float4 wv = *(const float4*)&W[(size_t)(e0 + er) * D_ + c];
      Ws[(c + 0) * XS + er] = (bf16_t)wv.x;
      Ws[(c + 1) * XS + er] = (bf16_t)wv.y;
      Ws[(c + 2) * XS + er] = (bf16_t)wv.z;
      Ws[(c + 3) * XS + er] = (bf16_t)wv.w;
    }
    __syncthreads();

    // A fragment (16x32): lane holds row (wave*16+lrow), K chunks {8g..8g+7, 16+8g..}
    v16bf a;
    {
      const bf16_t* rp = &Xs[(wave * 16 + lrow) * XS + 8 * g];
      v8bf lo = *(const v8bf*)&rp[0];
      v8bf hi = *(const v8bf*)&rp[16];
      for (int i = 0; i < 8; ++i) { a[i] = lo[i]; a[8 + i] = hi[i]; }
    }
    // 8 N-tiles of 16: B fragment lane holds col nt*16+lrow, K chunks {8g..8g+7, 16+8g..}
    for (int nt = 0; nt < 8; ++nt) {
      const bf16_t* rb = &Ws[(nt * 16 + lrow) * XS + 8 * g];
      v8bf lo = *(const v8bf*)&rb[0];
      v8bf hi = *(const v8bf*)&rb[16];
      v16bf bf;
      for (int i = 0; i < 8; ++i) { bf[i] = lo[i]; bf[8 + i] = hi[i]; }
      acc[nt] = __builtin_amdgcn_wmma_f32_16x16x32_bf16(
          false, a, false, bf, (short)0, acc[nt], false, false);
    }
    __syncthreads();
  }

  // Epilogue: C/D layout row M = r + 8g, col N = lane%16
  const int mrow_base = m0 + wave * 16 + 8 * g;
  if (z < 2) {
    bf16_t* out = (z == 0) ? Q : Kb;
    for (int nt = 0; nt < 8; ++nt)
      for (int r = 0; r < 8; ++r)
        out[(size_t)(mrow_base + r) * D_ + nt * 16 + lrow] = (bf16_t)acc[nt][r];
  } else {
    const int b = m0 / S_;                     // 128-row tile never crosses a batch
    for (int nt = 0; nt < 8; ++nt)
      for (int r = 0; r < 8; ++r) {
        int srow = (mrow_base + r) - b * S_;
        Vt[(size_t)b * D_ * S_ + (size_t)(nt * 16 + lrow) * S_ + srow] =
            (bf16_t)acc[nt][r];
      }
  }
}

// ---------------------------------------------------------------------------
// Kernel 2: single-pass flash attention, one wave = 16 query rows.
// K/V key tiles staged once per block via the Tensor Data Mover into LDS and
// shared by all 8 waves.  S = softmax(Q K^T/sqrt(D) - 1e15*mask); out = S V.
// ---------------------------------------------------------------------------
__launch_bounds__(256)
__global__ void attn(const bf16_t* __restrict__ Q,
                     const bf16_t* __restrict__ Kb,
                     const bf16_t* __restrict__ Vt,
                     const int* __restrict__ mask,
                     float* __restrict__ out) {
  constexpr int PS = 40;                               // padded LDS stride
  __shared__ __align__(64) bf16_t Klds[32 * D_];       // [key][d]  8 KB
  __shared__ __align__(64) bf16_t Vlds[D_ * 32];       // [d][key]  8 KB
  __shared__ __align__(32) bf16_t Plds[8][16 * PS];    // per-wave P scratch

  const int tid  = threadIdx.x;
  const int wave = tid >> 5;
  const int lane = tid & 31;
  const int lrow = lane & 15;
  const int g    = lane >> 4;

  const int q0 = blockIdx.x * 128 + wave * 16;  // global query row base
  const int b  = q0 / S_;
  const int qs = q0 - b * S_;

  // Preload Q A-fragments for all 4 d-chunks of 32
  v16bf aq[4];
  for (int c = 0; c < 4; ++c) {
    const bf16_t* base = &Q[(size_t)(q0 + lrow) * D_ + c * 32 + 16 * g];
    v8bf lo = *(const v8bf*)&base[0];
    v8bf hi = *(const v8bf*)&base[8];
    for (int i = 0; i < 8; ++i) { aq[c][i] = lo[i]; aq[c][8 + i] = hi[i]; }
  }

  float m_run[8], l_run[8];
  v8f o[8] = {};
  for (int r = 0; r < 8; ++r) { m_run[r] = -INFINITY; l_run[r] = 0.0f; }

  const float scale = 0.08838834764831843f;     // 1/sqrt(128)
  const int* mbase = &mask[(size_t)b * S_ * S_];
  bf16_t* pw = &Plds[wave][0];

  const u32 klds_addr = (u32)(uintptr_t)&Klds[0];
  const u32 vlds_addr = (u32)(uintptr_t)&Vlds[0];
  const u64 kgbase = (u64)(uintptr_t)&Kb[(size_t)b * S_ * D_];
  const u64 vgbase = (u64)(uintptr_t)&Vt[(size_t)b * D_ * S_];

  for (int k0 = 0; k0 < S_; k0 += 32) {
    // ---- stage K/V key tiles into LDS via the Tensor Data Mover ----
    if (wave == 0) {
      // K tile: rows k0..k0+31 of K[b] ([S][D] bf16) -> Klds [32][128]
      tdm_load_2d(kgbase + (u64)k0 * D_ * 2, klds_addr,
                  /*dim0=*/D_, /*dim1=*/32, /*stride0=*/D_,
                  /*tile0=*/D_, /*tile1=*/32);
      // V tile: cols k0..k0+31 of Vt[b] ([D][S] bf16) -> Vlds [128][32]
      tdm_load_2d(vgbase + (u64)k0 * 2, vlds_addr,
                  /*dim0=*/32, /*dim1=*/D_, /*stride0=*/S_,
                  /*tile0=*/32, /*tile1=*/D_);
      __builtin_amdgcn_s_wait_tensorcnt(0);
    }
    __syncthreads();

    // Prefetch next mask tile (dominant HBM stream, 67 MB)
    if (k0 + 32 < S_)
      __builtin_prefetch(&mbase[(size_t)(qs + 8 * g) * S_ + k0 + 32], 0, 1);

    // ---- scores: two 16-key subtiles, contract over d=128 (4 WMMAs each) ----
    v8f s0 = {}, s1 = {};
    for (int c = 0; c < 4; ++c) {
      v16bf f0 = *(const v16bf*)&Klds[(size_t)lrow * D_ + c * 32 + 16 * g];
      v16bf f1 = *(const v16bf*)&Klds[(size_t)(16 + lrow) * D_ + c * 32 + 16 * g];
      s0 = __builtin_amdgcn_wmma_f32_16x16x32_bf16(false, aq[c], false, f0, (short)0, s0, false, false);
      s1 = __builtin_amdgcn_wmma_f32_16x16x32_bf16(false, aq[c], false, f1, (short)0, s1, false, false);
    }

    // ---- mask, online softmax (row lives in 16 lanes of one wave half) ----
    float p0[8], p1[8];
    for (int r = 0; r < 8; ++r) {
      const int qrow = qs + r + 8 * g;
      const int* mr = &mbase[(size_t)qrow * S_ + k0];
      float v0 = s0[r] * scale - 1e15f * (float)mr[lrow];
      float v1 = s1[r] * scale - 1e15f * (float)mr[16 + lrow];
      float t = fmaxf(v0, v1);
      for (int x = 1; x < 16; x <<= 1) t = fmaxf(t, __shfl_xor(t, x, 16));
      float nm   = fmaxf(m_run[r], t);
      float corr = __expf(m_run[r] - nm);
      p0[r] = __expf(v0 - nm);
      p1[r] = __expf(v1 - nm);
      float sum = p0[r] + p1[r];
      for (int x = 1; x < 16; x <<= 1) sum += __shfl_xor(sum, x, 16);
      l_run[r] = l_run[r] * corr + sum;
      m_run[r] = nm;
      for (int dt = 0; dt < 8; ++dt) o[dt][r] *= corr;
    }

    // ---- re-layout P (C/D layout -> A layout) via per-wave LDS ----
    for (int r = 0; r < 8; ++r) {
      pw[(r + 8 * g) * PS + lrow]      = (bf16_t)p0[r];
      pw[(r + 8 * g) * PS + 16 + lrow] = (bf16_t)p1[r];
    }
    v16bf ap;
    {
      const bf16_t* pr = &pw[lrow * PS + 16 * g];
      v8bf lo = *(const v8bf*)&pr[0];
      v8bf hi = *(const v8bf*)&pr[8];
      for (int i = 0; i < 8; ++i) { ap[i] = lo[i]; ap[8 + i] = hi[i]; }
    }

    // ---- O += P * V  (V tile in LDS, B-operand contiguous in keys) ----
    for (int dt = 0; dt < 8; ++dt) {
      v16bf f = *(const v16bf*)&Vlds[(size_t)(dt * 16 + lrow) * 32 + 16 * g];
      o[dt] = __builtin_amdgcn_wmma_f32_16x16x32_bf16(false, ap, false, f, (short)0, o[dt], false, false);
    }
    __syncthreads();   // all waves done reading before next TDM overwrite
  }

  // ---- normalize and write f32 output ----
  for (int r = 0; r < 8; ++r) {
    const float inv = 1.0f / l_run[r];
    for (int dt = 0; dt < 8; ++dt)
      out[(size_t)(q0 + r + 8 * g) * D_ + dt * 16 + lrow] = o[dt][r] * inv;
  }
}

// ---------------------------------------------------------------------------
extern "C" void kernel_launch(void* const* d_in, const int* in_sizes, int n_in,
                              void* d_out, int out_size, void* d_ws, size_t ws_size,
                              hipStream_t stream) {
  const float* X    = (const float*)d_in[0];
  const int*   mask = (const int*)d_in[1];
  const float* Wq   = (const float*)d_in[2];
  const float* Wk   = (const float*)d_in[3];
  const float* Wv   = (const float*)d_in[4];
  float* out = (float*)d_out;

  // Workspace: Q (2 MB) | K (2 MB) | Vt (2 MB), all bf16
  char* ws = (char*)d_ws;
  bf16_t* Qb = (bf16_t*)(ws);
  bf16_t* Kb = (bf16_t*)(ws + (size_t)M_ * D_ * 2);
  bf16_t* Vt = (bf16_t*)(ws + (size_t)M_ * D_ * 4);

  dim3 grid1(M_ / 128, 1, 3);
  qkv_proj<<<grid1, dim3(256), 0, stream>>>(X, Wq, Wk, Wv, Qb, Kb, Vt);
  attn<<<dim3(M_ / 128), dim3(256), 0, stream>>>(Qb, Kb, Vt, mask, out);
}
